// GeoVMRNN_Supervised_36172214567659
// MI455X (gfx1250) — matrix operand order repaired and checked by reading, not compiled, not run on hardware
//
#include <hip/hip_runtime.h>
#include <hip/hip_bf16.h>
#include <cstdint>

typedef __bf16 bf16;
typedef __attribute__((ext_vector_type(8)))  bf16  v8bf;
typedef __attribute__((ext_vector_type(16))) bf16  v16bf;
typedef __attribute__((ext_vector_type(8)))  float v8f;

constexpr int cB = 4, cT = 6, cC = 4, cH = 32, cW = 32, cL = 1024;
constexpr int cD = 256, cDI = 512, cN = 16, cR = 16, cK = 4;
constexpr int cM = cB * cL;        // 4096 tokens per timestep
constexpr int cBT = cB * cT;       // 24 frames

// ---------------------------------------------------------------------------
// WMMA GEMM:  C[M,N] (f32) = A[M,K] (bf16, lda) * W[N,K]^T (bf16, ldb) + bias
// block = 256 threads (8 wave32), 64x64 tile, K chunks of 32.
// Full tiles: async global->LDS DMA (b128/thread, ASYNCcnt).
// Ragged tiles: guarded b128 vector loads.
// All lda/ldb/K in this model are multiples of 8 elems (16B), buffers 256B
// aligned, so 16B vector transactions are always legal.
// ---------------------------------------------------------------------------
#define LDSP 48  // padded LDS row stride in bf16 elems (96B, 16B aligned)

__global__ __launch_bounds__(256) void k_gemm_bf16(
    const bf16* __restrict__ A, int lda,
    const bf16* __restrict__ Bw, int ldb,
    float* __restrict__ C, int ldc,
    const float* __restrict__ bias,
    int M, int Ndim, int Kd, int acc)
{
  __shared__ bf16 As[64 * LDSP];
  __shared__ bf16 Bs[64 * LDSP];
  const int tid = threadIdx.x;
  const int lid = tid & 31;
  const int wid = tid >> 5;
  const int wm  = wid & 3;   // wave M tile 0..3
  const int wn  = wid >> 2;  // wave N tile 0..1 (32 cols each)
  const int m_base = blockIdx.y * 64;
  const int n_base = blockIdx.x * 64;
  const int lrow = tid >> 2;         // 0..63
  const int lcol = (tid & 3) * 8;    // 0,8,16,24
  const bool mn_full = (m_base + 64 <= M) && (n_base + 64 <= Ndim);

  const int gr = m_base + lrow;
  const int gn = n_base + lrow;
  bf16* lap = &As[lrow * LDSP + lcol];
  bf16* lbp = &Bs[lrow * LDSP + lcol];
  // wave-relative LDS byte offset = low 32 bits of the generic address
  const unsigned la = (unsigned)(uintptr_t)lap;
  const unsigned lb = (unsigned)(uintptr_t)lbp;

  v8f c0 = {}; v8f c1 = {};

  for (int k0 = 0; k0 < Kd; k0 += 32) {
    const bf16* gap = A  + (size_t)gr * lda + (k0 + lcol);
    const bf16* gbp = Bw + (size_t)gn * ldb + (k0 + lcol);
    if (mn_full && (k0 + 32 <= Kd)) {
      // CDNA5 Tensor/async path: DMA 16B per lane straight into LDS.
      asm volatile("global_load_async_to_lds_b128 %0, %1, off"
                   :: "v"(la), "v"(gap) : "memory");
      asm volatile("global_load_async_to_lds_b128 %0, %1, off"
                   :: "v"(lb), "v"(gbp) : "memory");
      if (k0 + 64 <= Kd) {  // pull next K chunk toward L2
        __builtin_prefetch(gap + 32, 0, 0);
        __builtin_prefetch(gbp + 32, 0, 0);
      }
      asm volatile("s_wait_asynccnt 0x0" ::: "memory");
    } else {
      v8bf av = {}, bv = {};
      if (gr < M    && (k0 + lcol + 8 <= Kd)) av = *(const v8bf*)gap;
      if (gn < Ndim && (k0 + lcol + 8 <= Kd)) bv = *(const v8bf*)gbp;
      *(v8bf*)lap = av;
      *(v8bf*)lbp = bv;
    }
    __syncthreads();

    // A fragment: ISA 16-bit 16x32 layout. lanes<16: K0-7 / K16-23 ; lanes>=16: K8-15 / K24-31
    const int arow = wm * 16 + (lid & 15);
    const int aklo = (lid < 16) ? 0 : 8;
    v8bf alo = *(const v8bf*)&As[arow * LDSP + aklo];
    v8bf ahi = *(const v8bf*)&As[arow * LDSP + aklo + 16];
    v16bf af = __builtin_shufflevector(alo, ahi, 0,1,2,3,4,5,6,7,8,9,10,11,12,13,14,15);

    // B fragments (N-major staged, lane = N, lanes<16: K0-7 / K16-23, lanes>=16: K8-15 / K24-31)
    const int brow0 = wn * 32 + (lid & 15);
    const int bko = (lid < 16) ? 0 : 8;
    v8bf b0lo = *(const v8bf*)&Bs[brow0 * LDSP + bko];
    v8bf b0hi = *(const v8bf*)&Bs[brow0 * LDSP + bko + 16];
    v16bf bf0 = __builtin_shufflevector(b0lo, b0hi, 0,1,2,3,4,5,6,7,8,9,10,11,12,13,14,15);
    v8bf b1lo = *(const v8bf*)&Bs[(brow0 + 16) * LDSP + bko];
    v8bf b1hi = *(const v8bf*)&Bs[(brow0 + 16) * LDSP + bko + 16];
    v16bf bf1 = __builtin_shufflevector(b1lo, b1hi, 0,1,2,3,4,5,6,7,8,9,10,11,12,13,14,15);

    c0 = __builtin_amdgcn_wmma_f32_16x16x32_bf16(false, af, false, bf0, (short)0, c0, false, false);
    c1 = __builtin_amdgcn_wmma_f32_16x16x32_bf16(false, af, false, bf1, (short)0, c1, false, false);
    __syncthreads();
  }

  // C/D layout: lane(N), VGPR r -> M = r (+8 for lanes 16-31)
  const int nc0 = n_base + wn * 32 + (lid & 15);
  const int mrow = m_base + wm * 16 + ((lid >> 4) << 3);
#pragma unroll
  for (int r = 0; r < 8; ++r) {
    const int m = mrow + r;
    if (m >= M) continue;
    if (nc0 < Ndim) {
      float v = c0[r] + (bias ? bias[nc0] : 0.0f);
      if (acc) v += C[(size_t)m * ldc + nc0];
      C[(size_t)m * ldc + nc0] = v;
    }
    const int nc1 = nc0 + 16;
    if (nc1 < Ndim) {
      float v = c1[r] + (bias ? bias[nc1] : 0.0f);
      if (acc) v += C[(size_t)m * ldc + nc1];
      C[(size_t)m * ldc + nc1] = v;
    }
  }
}

// ---------------------------------------------------------------------------
// Small helper kernels
// ---------------------------------------------------------------------------
__global__ void k_zero(float* p, int n) {
  int i = blockIdx.x * blockDim.x + threadIdx.x;
  if (i < n) p[i] = 0.0f;
}

__global__ void k_f32_bf16(const float* __restrict__ in, bf16* __restrict__ out, int n) {
  int i = blockIdx.x * blockDim.x + threadIdx.x;
  if (i < n) out[i] = (bf16)in[i];
}

// in[K][N] (f32) -> out[N][K] (bf16)   (weight -> N-major)
__global__ void k_transpose_f32_bf16(const float* __restrict__ in, bf16* __restrict__ out,
                                     int K, int N) {
  int i = blockIdx.x * blockDim.x + threadIdx.x;
  if (i >= K * N) return;
  int k = i / N, n = i % N;
  out[(size_t)n * K + k] = (bf16)in[i];
}

// ---------------------------------------------------------------------------
// Convolutions (direct, f32)
// ---------------------------------------------------------------------------
__global__ void k_conv3x3(const float* __restrict__ in, const float* __restrict__ w,
                          const float* __restrict__ bias,
                          const float* bng, const float* bnb, const float* bnm, const float* bnv,
                          float* __restrict__ out, int NB, int Cin, int Cout, int relu) {
  int idx = blockIdx.x * blockDim.x + threadIdx.x;
  if (idx >= NB * Cout * cL) return;
  int pix = idx % cL;
  int co  = (idx / cL) % Cout;
  int n   = idx / (cL * Cout);
  int h = pix >> 5, x = pix & 31;
  float acc = bias[co];
  const float* inb = in + (size_t)n * Cin * cL;
  const float* wb  = w + (size_t)co * Cin * 9;
  for (int ci = 0; ci < Cin; ++ci) {
    const float* ip = inb + ci * cL;
    const float* wp = wb + ci * 9;
#pragma unroll
    for (int kh = 0; kh < 3; ++kh) {
      int ih = h + kh - 1;
      if (ih < 0 || ih >= cH) continue;
#pragma unroll
      for (int kw = 0; kw < 3; ++kw) {
        int iw = x + kw - 1;
        if (iw < 0 || iw >= cW) continue;
        acc += ip[ih * cW + iw] * wp[kh * 3 + kw];
      }
    }
  }
  if (bng) acc = bng[co] * (acc - bnm[co]) * rsqrtf(bnv[co] + 1e-5f) + bnb[co];
  if (relu) acc = fmaxf(acc, 0.0f);
  out[idx] = acc;
}

__global__ void k_conv1x1(const float* __restrict__ in, const float* __restrict__ w,
                          const float* __restrict__ bias,
                          const float* bng, const float* bnb, const float* bnm, const float* bnv,
                          float* __restrict__ out, int NB, int Cin, int Cout, int relu, int wt) {
  int idx = blockIdx.x * blockDim.x + threadIdx.x;
  if (idx >= NB * Cout * cL) return;
  int pix = idx % cL;
  int co  = (idx / cL) % Cout;
  int n   = idx / (cL * Cout);
  float acc = bias ? bias[co] : 0.0f;
  const float* ip = in + (size_t)n * Cin * cL + pix;
  for (int ci = 0; ci < Cin; ++ci)
    acc += ip[(size_t)ci * cL] * (wt ? w[(size_t)ci * Cout + co] : w[(size_t)co * Cin + ci]);
  if (bng) acc = bng[co] * (acc - bnm[co]) * rsqrtf(bnv[co] + 1e-5f) + bnb[co];
  if (relu) acc = fmaxf(acc, 0.0f);
  out[idx] = acc;
}

// spatial attention: x *= sigmoid(conv1x1_{256->1}(x)), in-place, per pixel
__global__ void k_satt(float* __restrict__ x, const float* __restrict__ saw,
                       const float* __restrict__ sab, int NB) {
  int idx = blockIdx.x * blockDim.x + threadIdx.x;
  if (idx >= NB * cL) return;
  int pix = idx % cL, n = idx / cL;
  float* xb = x + (size_t)n * cD * cL + pix;
  float s = sab[0];
  for (int c = 0; c < cD; ++c) s += xb[(size_t)c * cL] * saw[c];
  float a = 1.0f / (1.0f + expf(-s));
  for (int c = 0; c < cD; ++c) xb[(size_t)c * cL] *= a;
}

// depthwise 3x3 + bias + silu on [B, DI, 32, 32]
__global__ void k_dwconv_silu(const float* __restrict__ in, const float* __restrict__ w,
                              const float* __restrict__ bias, float* __restrict__ out) {
  int idx = blockIdx.x * blockDim.x + threadIdx.x;
  if (idx >= cB * cDI * cL) return;
  int pix = idx % cL;
  int d   = (idx / cL) % cDI;
  int h = pix >> 5, x = pix & 31;
  const float* ip = in + (size_t)(idx - pix);
  const float* wp = w + d * 9;
  float acc = bias[d];
#pragma unroll
  for (int kh = 0; kh < 3; ++kh) {
    int ih = h + kh - 1;
    if (ih < 0 || ih >= cH) continue;
#pragma unroll
    for (int kw = 0; kw < 3; ++kw) {
      int iw = x + kw - 1;
      if (iw < 0 || iw >= cW) continue;
      acc += ip[ih * cW + iw] * wp[kh * 3 + kw];
    }
  }
  out[idx] = acc / (1.0f + expf(-acc));   // silu
}

// ---------------------------------------------------------------------------
// SS2D data movement + scan
// ---------------------------------------------------------------------------
// xz[M,1024] first half -> xi[B,DI,H,W]
__global__ void k_xz_to_img(const float* __restrict__ xz, float* __restrict__ xi) {
  int idx = blockIdx.x * blockDim.x + threadIdx.x;
  if (idx >= cB * cDI * cL) return;
  int pix = idx % cL;
  int d   = (idx / cL) % cDI;
  int b   = idx / (cL * cDI);
  xi[idx] = xz[((size_t)b * cL + pix) * (2 * cDI) + d];
}

// xi[B,DI,L] -> xs[B,4,DI,L] : hw, wh, flip(hw), flip(wh)
__global__ void k_build_xs(const float* __restrict__ xi, float* __restrict__ xs) {
  int idx = blockIdx.x * blockDim.x + threadIdx.x;
  if (idx >= cB * cDI * cL) return;
  int l = idx % cL;
  int d = (idx / cL) % cDI;
  int b = idx / (cL * cDI);
  int h = l >> 5, w = l & 31;
  int lt = (w << 5) | h;
  float v = xi[idx];
  size_t base = (((size_t)b * cK) * cDI + d) * cL;
  size_t sk = (size_t)cDI * cL;
  xs[base + l] = v;
  xs[base + sk + lt] = v;
  xs[base + 2 * sk + (cL - 1 - l)] = v;
  xs[base + 3 * sk + (cL - 1 - lt)] = v;
}

// xs[B,4,DI,L] -> A matrices axs[k][(b*L+l)][DI] (bf16) for x_proj GEMMs
__global__ void k_xs_to_bf16(const float* __restrict__ xs, bf16* __restrict__ axs) {
  int idx = blockIdx.x * blockDim.x + threadIdx.x;
  if (idx >= cB * cK * cDI * cL) return;
  int l = idx % cL;
  int d = (idx / cL) % cDI;
  int k = (idx / (cL * cDI)) % cK;
  int b = idx / (cL * cDI * cK);
  axs[((size_t)k * cM + (size_t)b * cL + l) * cDI + d] = (bf16)xs[idx];
}

// one thread per (b,k,d): serial recurrence over L with 16-wide state in regs
__global__ void k_scan(const float* __restrict__ xs, const float* __restrict__ dts,
                       const float* __restrict__ xdbl, const float* __restrict__ a_log,
                       const float* __restrict__ dt_b, const float* __restrict__ Ds,
                       float* __restrict__ oy) {
  int idx = blockIdx.x * blockDim.x + threadIdx.x;
  if (idx >= cB * cK * cDI) return;
  int d = idx % cDI;
  int k = (idx / cDI) % cK;
  int b = idx / (cDI * cK);
  float A[cN], h[cN];
#pragma unroll
  for (int n = 0; n < cN; ++n) {
    A[n] = -expf(a_log[((size_t)k * cDI + d) * cN + n]);
    h[n] = 0.0f;
  }
  float dtb = dt_b[k * cDI + d];
  float Dv  = Ds[k * cDI + d];
  size_t xbase = (((size_t)b * cK + k) * cDI + d) * cL;
  size_t mbase = (size_t)k * cM + (size_t)b * cL;
  for (int l = 0; l < cL; ++l) {
    float draw = dts[(mbase + l) * cDI + d] + dtb;
    float delta = draw > 20.0f ? draw : log1pf(expf(draw));
    float xv = xs[xbase + l];
    const float* row = xdbl + (mbase + l) * 48;
    float dx = delta * xv;
    float y = 0.0f;
#pragma unroll
    for (int n = 0; n < cN; ++n) {
      h[n] = expf(delta * A[n]) * h[n] + dx * row[cR + n];
      y += h[n] * row[cR + cN + n];
    }
    oy[xbase + l] = y + Dv * xv;
  }
}

// merge the 4 scan directions -> ym[(b*L+l)][DI]
__global__ void k_merge_y(const float* __restrict__ oy, float* __restrict__ ym) {
  int idx = blockIdx.x * blockDim.x + threadIdx.x;
  if (idx >= cB * cL * cDI) return;
  int d = idx % cDI;
  int l = (idx / cDI) % cL;
  int b = idx / (cDI * cL);
  int h = l >> 5, w = l & 31;
  int lt = (w << 5) | h;
  size_t base = ((size_t)b * cK) * cDI * cL + (size_t)d * cL;
  size_t sk = (size_t)cDI * cL;
  ym[idx] = oy[base + l] + oy[base + 2 * sk + (cL - 1 - l)]
          + oy[base + sk + lt] + oy[base + 3 * sk + (cL - 1 - lt)];
}

// ---------------------------------------------------------------------------
// LayerNorms / gating / cell
// ---------------------------------------------------------------------------
// LN over last dim (Cdim), output bf16 at [row*ostride + ocol + c]
__global__ void k_ln_bf16(const float* __restrict__ x, const float* __restrict__ g,
                          const float* __restrict__ bb, bf16* __restrict__ out,
                          int Cdim, int ostride, int ocol, float eps, int Mrows) {
  int row = blockIdx.x * blockDim.x + threadIdx.x;
  if (row >= Mrows) return;
  const float* xr = x + (size_t)row * Cdim;
  float mu = 0.0f;
  for (int c2 = 0; c2 < Cdim; ++c2) mu += xr[c2];
  mu /= Cdim;
  float var = 0.0f;
  for (int c2 = 0; c2 < Cdim; ++c2) { float t = xr[c2] - mu; var += t * t; }
  var /= Cdim;
  float rs = rsqrtf(var + eps);
  bf16* o = out + (size_t)row * ostride + ocol;
  for (int c2 = 0; c2 < Cdim; ++c2)
    o[c2] = (bf16)((xr[c2] - mu) * rs * g[c2] + bb[c2]);
}

// out_norm LN(512) then *= silu(z) (z = xz[:,512:]); write bf16 for out_proj A
__global__ void k_ln_gate(const float* __restrict__ ym, const float* __restrict__ xz,
                          const float* __restrict__ g, const float* __restrict__ bb,
                          bf16* __restrict__ out, int Mrows) {
  int row = blockIdx.x * blockDim.x + threadIdx.x;
  if (row >= Mrows) return;
  const float* yr = ym + (size_t)row * cDI;
  const float* zr = xz + (size_t)row * (2 * cDI) + cDI;
  float mu = 0.0f;
  for (int c2 = 0; c2 < cDI; ++c2) mu += yr[c2];
  mu /= cDI;
  float var = 0.0f;
  for (int c2 = 0; c2 < cDI; ++c2) { float t = yr[c2] - mu; var += t * t; }
  var /= cDI;
  float rs = rsqrtf(var + 1e-5f);
  bf16* o = out + (size_t)row * cDI;
  for (int c2 = 0; c2 < cDI; ++c2) {
    float yn = (yr[c2] - mu) * rs * g[c2] + bb[c2];
    float z = zr[c2];
    o[c2] = (bf16)(yn * z / (1.0f + expf(-z)));
  }
}

// LSTM-ish cell: Ft=sig(o); Ct=Ft*(cx+tanh(o)); Ht=Ft*tanh(Ct)
__global__ void k_cell(const float* __restrict__ o, float* __restrict__ hbuf,
                       float* __restrict__ cbuf, float* __restrict__ Hs) {
  int idx = blockIdx.x * blockDim.x + threadIdx.x;
  if (idx >= cM * cD) return;
  float ov = o[idx];
  float Ft = 1.0f / (1.0f + expf(-ov));
  float Ct = Ft * (cbuf[idx] + tanhf(ov));
  float Ht = Ft * tanhf(Ct);
  cbuf[idx] = Ct;
  hbuf[idx] = Ht;
  if (Hs) Hs[idx] = Ht;
}

// ---------------------------------------------------------------------------
// Framing / token reshapes
// ---------------------------------------------------------------------------
__global__ void k_connect(const float* __restrict__ predictor,
                          const float* __restrict__ predictand, float* __restrict__ out) {
  int idx = blockIdx.x * blockDim.x + threadIdx.x;
  if (idx >= cB * cT * cC * cL) return;
  int e = idx % (cC * cL);
  int t = (idx / (cC * cL)) % cT;
  int b = idx / (cC * cL * cT);
  out[idx] = (t == 0) ? predictor[((size_t)b * cT + (cT - 1)) * cC * cL + e]
                      : predictand[((size_t)b * cT + (t - 1)) * cC * cL + e];
}

// pf[(b*T+t)][D][L] -> tok[t][b][l][d]
__global__ void k_tokens(const float* __restrict__ pf, float* __restrict__ tok) {
  int idx = blockIdx.x * blockDim.x + threadIdx.x;
  if (idx >= cT * cB * cL * cD) return;
  int d = idx % cD;
  int l = (idx / cD) % cL;
  int b = (idx / (cD * cL)) % cB;
  int t = idx / (cD * cL * cB);
  tok[idx] = pf[(((size_t)b * cT + t) * cD + d) * cL + l];
}

// Hs2[t][b][l][d] -> img[(b*T+t)][d][l]
__global__ void k_tok_to_img(const float* __restrict__ Hs, float* __restrict__ img) {
  int idx = blockIdx.x * blockDim.x + threadIdx.x;
  if (idx >= cBT * cD * cL) return;
  int pix = idx % cL;
  int d   = (idx / cL) % cD;
  int n   = idx / (cL * cD);
  int b = n / cT, t = n % cT;
  img[idx] = Hs[(((size_t)t * cB + b) * cL + pix) * cD + d];
}

// channel concat [24,256,L] + [24,256,L] -> [24,512,L]
__global__ void k_concat_ch(const float* __restrict__ g2, const float* __restrict__ vm,
                            float* __restrict__ out) {
  int idx = blockIdx.x * blockDim.x + threadIdx.x;
  if (idx >= cBT * 2 * cD * cL) return;
  int pix = idx % cL;
  int ch  = (idx / cL) % (2 * cD);
  int n   = idx / (cL * 2 * cD);
  out[idx] = (ch < cD) ? g2[((size_t)n * cD + ch) * cL + pix]
                       : vm[((size_t)n * cD + (ch - cD)) * cL + pix];
}

// ---------------------------------------------------------------------------
// Parameter-leaf indexing (alphabetical flatten within nested param dicts)
// ---------------------------------------------------------------------------
struct SS2DIdx { int A_log, Ds, conv_b, conv_w, dt_b, dt_w, in_proj, out_norm_b, out_norm_g, out_proj, x_proj; };
struct VSBIdx  { int lin_b, lin_w, ln1_b, ln1_g; SS2DIdx ss; };
struct BNIdx   { int b, g, m, v; };

static SS2DIdx takeSS(int& c) {
  SS2DIdx s;
  s.A_log = c++; s.Ds = c++; s.conv_b = c++; s.conv_w = c++; s.dt_b = c++; s.dt_w = c++;
  s.in_proj = c++; s.out_norm_b = c++; s.out_norm_g = c++; s.out_proj = c++; s.x_proj = c++;
  return s;
}
static VSBIdx takeVSB(int& c) {
  VSBIdx v;
  v.lin_b = c++; v.lin_w = c++; v.ln1_b = c++; v.ln1_g = c++; v.ss = takeSS(c);
  return v;
}
static BNIdx takeBN(int& c) { BNIdx x; x.b = c++; x.g = c++; x.m = c++; x.v = c++; return x; }

#define EL(kern, total, ...) kern<<<dim3(((total) + 255) / 256), dim3(256), 0, stream>>>(__VA_ARGS__)

extern "C" void kernel_launch(void* const* d_in, const int* in_sizes, int n_in,
                              void* d_out, int out_size, void* d_ws, size_t ws_size,
                              hipStream_t stream) {
  (void)in_sizes; (void)n_in; (void)out_size; (void)ws_size;

  // --- parameter leaf indices ---
  int cu = 2;  // 0 = predictor, 1 = predictand
  VSBIdx Idec0 = takeVSB(cu), Idec1 = takeVSB(cu), Ienc0 = takeVSB(cu), Ienc1 = takeVSB(cu);
  int fus_b = cu++; BNIdx fus_bn = takeBN(cu); int fus_w = cu++;
  BNIdx bn1 = takeBN(cu), bn2 = takeBN(cu), bn3 = takeBN(cu);
  int c1_b = cu++, c1_w = cu++, c2_b = cu++, c2_w = cu++, c3_b = cu++, c3_w = cu++;
  int sa_b = cu++, sa_w = cu++;
  int ph1_b = cu++; BNIdx ph1_bn = takeBN(cu); int ph1_w = cu++;
  int ph2_b = cu++; BNIdx ph2_bn = takeBN(cu); int ph2_w = cu++;
  int ph3_b = cu++, ph3_w = cu++;
  int pp_b = cu++, pp_w = cu++, pti_b = cu++, pti_w = cu++;

  auto F = [&](int i) -> const float* { return (const float*)d_in[i]; };
  const float* predictor  = F(0);
  const float* predictand = F(1);
  float* out = (float*)d_out;

  // --- workspace arena ---
  char* wsb = (char*)d_ws;
  size_t off = 0;
  auto allocF = [&](size_t n) -> float* { float* p = (float*)(wsb + off); off += ((n * 4 + 255) / 256) * 256; return p; };
  auto allocB = [&](size_t n) -> bf16*  { bf16*  p = (bf16*) (wsb + off); off += ((n * 2 + 255) / 256) * 256; return p; };

  float* connect = allocF((size_t)cB * cT * cC * cL);
  float* geo1 = allocF((size_t)cBT * 64 * cL);
  float* geo2 = allocF((size_t)cBT * 128 * cL);
  float* geo3 = allocF((size_t)cBT * 256 * cL);   // g / g2 (post-attention)
  float* pf   = allocF((size_t)cBT * 256 * cL);   // also reused as dec_img
  float* tokE = allocF((size_t)cT * cM * cD);
  float* tokD = allocF((size_t)cT * cM * cD);
  float* hbuf = allocF((size_t)cM * cD);
  float* cbuf = allocF((size_t)cM * cD);
  float* Hs2  = allocF((size_t)cT * cM * cD);
  float* v1   = allocF((size_t)cM * cD);
  float* v2   = allocF((size_t)cM * cD);
  float* xn2  = allocF((size_t)cM * cD);
  float* xz   = allocF((size_t)cM * 2 * cDI);
  float* xi   = allocF((size_t)cB * cDI * cL);
  float* xi2  = allocF((size_t)cB * cDI * cL);
  float* xs   = allocF((size_t)cB * cK * cDI * cL);
  float* xdbl = allocF((size_t)cK * cM * 48);
  float* dts  = allocF((size_t)cK * cM * cDI);
  float* oy   = allocF((size_t)cB * cK * cDI * cL);
  float* ym   = allocF((size_t)cM * cDI);
  float* vmimg  = allocF((size_t)cBT * 256 * cL);
  float* fusedc = allocF((size_t)cBT * 512 * cL);
  float* fout   = allocF((size_t)cBT * 512 * cL);
  float* p1b    = allocF((size_t)cBT * 256 * cL);
  float* p2b    = allocF((size_t)cBT * 256 * cL);
  bf16* abf0 = allocB((size_t)cM * 2 * cD);
  bf16* abf1 = allocB((size_t)cM * cD);
  bf16* axs  = allocB((size_t)cK * cM * cDI);
  bf16* adbl = allocB((size_t)cK * cM * 48);
  bf16* gbf  = allocB((size_t)cM * cDI);

  // --- weight prep: f32 -> bf16, N-major ---
  struct VSBW { bf16 *lin, *inp, *xp, *dtw, *outp; };
  auto prep = [&](const VSBIdx& I) -> VSBW {
    VSBW w;
    w.lin  = allocB(256 * 512);
    w.inp  = allocB(1024 * 256);
    w.outp = allocB(256 * 512);
    w.xp   = allocB(4 * 48 * 512);
    w.dtw  = allocB(4 * 512 * 16);
    EL(k_transpose_f32_bf16, 512 * 256,  F(I.lin_w),      w.lin,  512, 256);
    EL(k_transpose_f32_bf16, 256 * 1024, F(I.ss.in_proj), w.inp,  256, 1024);
    EL(k_transpose_f32_bf16, 512 * 256,  F(I.ss.out_proj),w.outp, 512, 256);
    EL(k_f32_bf16, 4 * 48 * 512, F(I.ss.x_proj), w.xp,  4 * 48 * 512);
    EL(k_f32_bf16, 4 * 512 * 16, F(I.ss.dt_w),   w.dtw, 4 * 512 * 16);
    return w;
  };
  VSBW Wenc0 = prep(Ienc0), Wenc1 = prep(Ienc1), Wdec0 = prep(Idec0), Wdec1 = prep(Idec1);

  auto gemm = [&](const bf16* A, int lda, const bf16* Bw, int ldb, float* C, int ldc,
                  const float* bias, int Mx, int Nx, int Kx, int acc) {
    dim3 g((Nx + 63) / 64, (Mx + 63) / 64);
    k_gemm_bf16<<<g, dim3(256), 0, stream>>>(A, lda, Bw, ldb, C, ldc, bias, Mx, Nx, Kx, acc);
  };

  auto run_vsb = [&](const float* x_in, const float* hx, const VSBIdx& I, const VSBW& Wg,
                     float* x_out) {
    const bf16* Ain; int Alda;
    if (hx) {
      EL(k_ln_bf16, cM, x_in, F(I.ln1_g), F(I.ln1_b), abf0, cD, 2 * cD, 0,   1e-6f, cM);
      EL(k_ln_bf16, cM, hx,   F(I.ln1_g), F(I.ln1_b), abf0, cD, 2 * cD, cD,  1e-6f, cM);
      gemm(abf0, 2 * cD, Wg.lin, 2 * cD, xn2, cD, F(I.lin_b), cM, cD, 2 * cD, 0);
      EL(k_f32_bf16, cM * cD, xn2, abf1, cM * cD);
      Ain = abf1; Alda = cD;
    } else {
      EL(k_ln_bf16, cM, x_in, F(I.ln1_g), F(I.ln1_b), abf0, cD, cD, 0, 1e-6f, cM);
      Ain = abf0; Alda = cD;
    }
    gemm(Ain, Alda, Wg.inp, cD, xz, 2 * cDI, nullptr, cM, 2 * cDI, cD, 0);
    EL(k_xz_to_img, cB * cDI * cL, xz, xi);
    EL(k_dwconv_silu, cB * cDI * cL, xi, F(I.ss.conv_w), F(I.ss.conv_b), xi2);
    EL(k_build_xs, cB * cDI * cL, xi2, xs);
    EL(k_xs_to_bf16, cB * cK * cDI * cL, xs, axs);
    for (int k4 = 0; k4 < cK; ++k4)
      gemm(axs + (size_t)k4 * cM * cDI, cDI, Wg.xp + (size_t)k4 * 48 * cDI, cDI,
           xdbl + (size_t)k4 * cM * 48, 48, nullptr, cM, 48, cDI, 0);
    EL(k_f32_bf16, cK * cM * 48, xdbl, adbl, cK * cM * 48);
    for (int k4 = 0; k4 < cK; ++k4)
      gemm(adbl + (size_t)k4 * cM * 48, 48, Wg.dtw + (size_t)k4 * cDI * cR, cR,
           dts + (size_t)k4 * cM * cDI, cDI, nullptr, cM, cDI, cR, 0);
    EL(k_scan, cB * cK * cDI, xs, dts, xdbl, F(I.ss.A_log), F(I.ss.dt_b), F(I.ss.Ds), oy);
    EL(k_merge_y, cB * cL * cDI, oy, ym);
    EL(k_ln_gate, cM, ym, xz, F(I.ss.out_norm_g), F(I.ss.out_norm_b), gbf, cM);
    hipMemcpyAsync(x_out, x_in, (size_t)cM * cD * sizeof(float),
                   hipMemcpyDeviceToDevice, stream);
    gemm(gbf, cDI, Wg.outp, cDI, x_out, cD, nullptr, cM, cD, cDI, 1);  // residual
  };

  auto tokenize = [&](const float* frames, float* tokens) {
    EL(k_conv3x3, cBT * 64 * cL,  frames, F(c1_w), F(c1_b), F(bn1.g), F(bn1.b), F(bn1.m), F(bn1.v), geo1, cBT, cC, 64, 1);
    EL(k_conv3x3, cBT * 128 * cL, geo1,   F(c2_w), F(c2_b), F(bn2.g), F(bn2.b), F(bn2.m), F(bn2.v), geo2, cBT, 64, 128, 1);
    EL(k_conv3x3, cBT * 256 * cL, geo2,   F(c3_w), F(c3_b), F(bn3.g), F(bn3.b), F(bn3.m), F(bn3.v), geo3, cBT, 128, 256, 1);
    EL(k_satt, cBT * cL, geo3, F(sa_w), F(sa_b), cBT);
    EL(k_conv1x1, cBT * 256 * cL, geo3, F(pp_w), F(pp_b), nullptr, nullptr, nullptr, nullptr, pf, cBT, 256, 256, 0, 0);
    EL(k_tokens, cT * cM * cD, pf, tokens);
  };

  // ---- forward ----
  EL(k_zero, cM * cD, hbuf, cM * cD);
  EL(k_zero, cM * cD, cbuf, cM * cD);

  tokenize(predictor, tokE);
  for (int t = 0; t < cT; ++t) {
    const float* xt = tokE + (size_t)t * cM * cD;
    run_vsb(xt, hbuf, Ienc0, Wenc0, v1);
    run_vsb(v1, nullptr, Ienc1, Wenc1, v2);
    EL(k_cell, cM * cD, v2, hbuf, cbuf, (float*)nullptr);
  }

  EL(k_connect, cB * cT * cC * cL, predictor, predictand, connect);
  tokenize(connect, tokD);
  for (int t = 0; t < cT; ++t) {
    const float* xt = tokD + (size_t)t * cM * cD;
    run_vsb(xt, hbuf, Idec0, Wdec0, v1);
    run_vsb(v1, nullptr, Idec1, Wdec1, v2);
    EL(k_cell, cM * cD, v2, hbuf, cbuf, Hs2 + (size_t)t * cM * cD);
  }

  EL(k_tok_to_img, cBT * cD * cL, Hs2, pf);  // pf reused as dec_img
  EL(k_conv1x1, cBT * 256 * cL, pf, F(pti_w), F(pti_b), nullptr, nullptr, nullptr, nullptr,
     vmimg, cBT, 256, 256, 0, 1 /* transposed weight */);
  EL(k_concat_ch, cBT * 512 * cL, geo3, vmimg, fusedc);
  EL(k_conv1x1, cBT * 512 * cL, fusedc, F(fus_w), F(fus_b), F(fus_bn.g), F(fus_bn.b),
     F(fus_bn.m), F(fus_bn.v), fout, cBT, 512, 512, 1, 0);
  EL(k_conv3x3, cBT * 256 * cL, fout, F(ph1_w), F(ph1_b), F(ph1_bn.g), F(ph1_bn.b),
     F(ph1_bn.m), F(ph1_bn.v), p1b, cBT, 512, 256, 1);
  EL(k_conv3x3, cBT * 256 * cL, p1b, F(ph2_w), F(ph2_b), F(ph2_bn.g), F(ph2_bn.b),
     F(ph2_bn.m), F(ph2_bn.v), p2b, cBT, 256, 256, 1);
  EL(k_conv3x3, cBT * cC * cL, p2b, F(ph3_w), F(ph3_b), nullptr, nullptr, nullptr, nullptr,
     out, cBT, 256, cC, 0);
}
